// GNNExample_72430328479777
// MI455X (gfx1250) — matrix operand order repaired and checked by reading, not compiled, not run on hardware
//
#include <hip/hip_runtime.h>

#define N_NODES 50000
#define N_EDGES 200000
#define IN_DIM  512
#define HID     2000
#define LAT     128

typedef __attribute__((ext_vector_type(16))) __bf16 v16bf;
typedef __attribute__((ext_vector_type(8)))  float  v8f;
typedef int v4i_gcc __attribute__((__vector_size__(16)));

__device__ __forceinline__ unsigned short f2bf(float f) {
    unsigned int u = __float_as_uint(f);
    u += 0x7FFFu + ((u >> 16) & 1u);         // round-to-nearest-even
    return (unsigned short)(u >> 16);
}
__device__ __forceinline__ unsigned int pack2bf(float a, float b) {
    return (unsigned int)f2bf(a) | ((unsigned int)f2bf(b) << 16);
}

union FragBF { unsigned int u[8]; v16bf v; };

// ---------- CDNA5 async global->LDS copy (ASYNCcnt path) with safe fallback ----------
#if defined(__has_builtin)
#  if __has_builtin(__builtin_amdgcn_global_load_async_to_lds_b128)
#    define USE_ASYNC_LDS 1
#  endif
#endif
#ifndef USE_ASYNC_LDS
#  define USE_ASYNC_LDS 0
#endif

__device__ __forceinline__ void g2l_16B(const void* gsrc, void* ldst) {
#if USE_ASYNC_LDS
    __attribute__((address_space(1))) void* g1 =
        (__attribute__((address_space(1))) void*)const_cast<void*>(gsrc);
    __attribute__((address_space(3))) void* l3 =
        (__attribute__((address_space(3))) void*)ldst;
    __builtin_amdgcn_global_load_async_to_lds_b128(
        (__attribute__((address_space(1))) v4i_gcc*)g1,
        (__attribute__((address_space(3))) v4i_gcc*)l3, 0, 0);
#else
    *(uint4*)ldst = *(const uint4*)gsrc;
#endif
}

__device__ __forceinline__ void async_join() {
#if USE_ASYNC_LDS
#  if __has_builtin(__builtin_amdgcn_s_wait_asynccnt)
    __builtin_amdgcn_s_wait_asynccnt(0);
#  else
    asm volatile("s_wait_asynccnt 0x0" ::: "memory");
#  endif
#endif
}

// ---------------- degree / normalization ----------------
__global__ void deg_count_kernel(const int* __restrict__ col, float* __restrict__ deg, int E) {
    int e = blockIdx.x * blockDim.x + threadIdx.x;
    if (e < E) atomicAdd(&deg[col[e]], 1.0f);
}

__global__ void dis_kernel(float* __restrict__ deg, int n) {
    int i = blockIdx.x * blockDim.x + threadIdx.x;
    if (i < n) deg[i] = rsqrtf(deg[i] + 1.0f);   // +1 for the self loop; always > 0
}

// ---------------- self loop: dst[i][:] = dis[i]^2 * src[i][:] ----------------
__global__ void selfloop_kernel(const float* __restrict__ src, const float* __restrict__ dis,
                                float* __restrict__ dst, int nnodes, int F4) {
    int t = blockIdx.x * blockDim.x + threadIdx.x;
    int i = t / F4;
    int fc = (t - i * F4) * 4;
    if (i >= nnodes) return;
    float d = dis[i];
    float s = d * d;
    int F = F4 * 4;
    const float4 v = *(const float4*)(src + (size_t)i * F + fc);
    float4 o; o.x = s * v.x; o.y = s * v.y; o.z = s * v.z; o.w = s * v.w;
    *(float4*)(dst + (size_t)i * F + fc) = o;
}

// ---------------- edges: dst[col][:] += dis[row]*dis[col] * src[row][:] ----------------
__global__ void edge_scatter_kernel(const int* __restrict__ row, const int* __restrict__ col,
                                    const float* __restrict__ dis, const float* __restrict__ src,
                                    float* __restrict__ dst, int E, int F4) {
    int t = blockIdx.x * blockDim.x + threadIdx.x;
    int e = t / F4;
    int fc = (t - e * F4) * 4;
    if (e >= E) return;
    int r = row[e], c = col[e];
    float w = dis[r] * dis[c];
    int F = F4 * 4;
    const float4 v = *(const float4*)(src + (size_t)r * F + fc);
    float* p = dst + (size_t)c * F + fc;
    atomicAdd(p + 0, w * v.x);
    atomicAdd(p + 1, w * v.y);
    atomicAdd(p + 2, w * v.z);
    atomicAdd(p + 3, w * v.w);
}

// ---------------- finalize: out = sigmoid(out + b2) ----------------
__global__ void finalize_kernel(float* __restrict__ out, const float* __restrict__ bias,
                                int total, int F) {
    int t = blockIdx.x * blockDim.x + threadIdx.x;
    if (t >= total) return;
    float v = out[t] + bias[t % F];
    out[t] = 1.0f / (1.0f + __expf(-v));
}

// ---------------- f32 -> bf16 row-major (for GEMM A operand) ----------------
__global__ void cvt_bf16_kernel(const float* __restrict__ src, unsigned short* __restrict__ dst,
                                int total4) {
    int t = blockIdx.x * blockDim.x + threadIdx.x;
    if (t >= total4) return;
    const float4 v = *(const float4*)(src + (size_t)t * 4);
    uint2 o; o.x = pack2bf(v.x, v.y); o.y = pack2bf(v.z, v.w);
    *(uint2*)(dst + (size_t)t * 4) = o;
}

// ---------------- f32 [K][N] -> pair-packed bf16 u32 [(K/2)][N] (GEMM B operand) --------
// Bp[p*N + n] = {bf(B[2p][n]) , bf(B[2p+1][n])<<16}  == exactly one ISA B-fragment dword.
__global__ void pack_weight_kernel(const float* __restrict__ B, unsigned int* __restrict__ Bp,
                                   int P, int N) {
    int t = blockIdx.x * blockDim.x + threadIdx.x;
    if (t >= P * N) return;
    int p = t / N, n = t - p * N;
    float a = B[(size_t)(2 * p) * N + n];
    float b = B[(size_t)(2 * p + 1) * N + n];
    Bp[t] = pack2bf(a, b);
}

// ---------------- WMMA bf16 GEMM ----------------
// C[M,N] = act(A[M,K] @ B[K,N] + bias)
// A: bf16 row-major. Bp: pair-packed u32 [(K/2)][N]. Pure 16B global->LDS copies, no VALU.
#define BM 128
#define BN 128
#define BK 32

template<bool RELU_BIAS_BF16_OUT>
__launch_bounds__(256)
__global__ void gemm_wmma_kernel(const unsigned short* __restrict__ A,
                                 const unsigned int* __restrict__ Bp,
                                 const float* __restrict__ bias,
                                 void* __restrict__ Cptr,
                                 int M, int N, int K) {
    // A tile: [128 rows][16 k-pairs] packed u32 (k contiguous per row).
    __shared__ unsigned int As32[BM * (BK / 2)];
    // B tile: [16 k-pair rows][128 n] of packed u32 (same layout as Bp).
    __shared__ unsigned int Bs32[(BK / 2) * BN];

    const int tid  = threadIdx.x;
    const int lane = tid & 31;
    const int wave = tid >> 5;      // 0..7
    const int wm   = wave & 3;      // wave M offset = wm*32
    const int wn   = wave >> 2;     // wave N offset = wn*64
    const int half = lane >> 4;     // K-interleave select
    const int lrow = lane & 15;

    const int m0 = blockIdx.y * BM;
    const int n0 = blockIdx.x * BN;

    v8f acc[2][4];
#pragma unroll
    for (int i = 0; i < 2; ++i)
#pragma unroll
        for (int j = 0; j < 4; ++j)
#pragma unroll
            for (int r = 0; r < 8; ++r) acc[i][j][r] = 0.0f;

    const int P  = K >> 1;                    // total k-pairs
    const int KT = (K + BK - 1) / BK;
    for (int kt = 0; kt < KT; ++kt) {
        const int k0 = kt * BK;
        const int p0 = k0 >> 1;
        __syncthreads();
        // ---- A tile: 512 chunks of 16B (8 bf16) ----
#pragma unroll
        for (int p = 0; p < 2; ++p) {
            int c  = p * 256 + tid;          // 0..511
            int rt = c >> 2;                 // 0..127
            int kc = (c & 3) << 3;           // 0,8,16,24
            int gr = m0 + rt, gk = k0 + kc;
            void* l = &As32[rt * (BK / 2) + (kc >> 1)];
            if (gr < M && gk < K) {          // K % 8 == 0 -> whole-chunk guard
                g2l_16B(A + (size_t)gr * K + gk, l);
            } else {
                *(uint4*)l = make_uint4(0u, 0u, 0u, 0u);   // zero K-tail (correctness)
            }
        }
        // ---- B tile: 512 chunks of 16B (4 packed u32) ----
#pragma unroll
        for (int p = 0; p < 2; ++p) {
            int c  = p * 256 + tid;
            int pr = c >> 5;                 // 0..15 pair-row
            int nc = (c & 31) << 2;          // 0,4,...,124
            int gp = p0 + pr, gn = n0 + nc;
            void* l = &Bs32[pr * BN + nc];
            if (gp < P && gn < N) {          // N % 4 == 0 -> whole-chunk guard
                g2l_16B(Bp + (size_t)gp * N + gn, l);
            } else {
                *(uint4*)l = make_uint4(0u, 0u, 0u, 0u);
            }
        }
        async_join();
        __syncthreads();

        // ---- fragments per ISA VGPR layouts, 8 WMMAs / wave / k-step ----
        FragBF a[2], b[4];
#pragma unroll
        for (int i = 0; i < 2; ++i) {
            int m = wm * 32 + i * 16 + lrow;
            const unsigned int* arow = &As32[m * (BK / 2)];
#pragma unroll
            for (int j = 0; j < 4; ++j) {
                a[i].u[j]     = arow[j + half * 4];        // K 0..7  / 8..15
                a[i].u[j + 4] = arow[8 + j + half * 4];    // K 16..23 / 24..31
            }
        }
#pragma unroll
        for (int j = 0; j < 4; ++j) {
            int n = wn * 64 + j * 16 + lrow;
#pragma unroll
            for (int q = 0; q < 8; ++q)
                b[j].u[q] = Bs32[(q + (q & 4) + half * 4) * BN + n]; // pairs 0..3,8..11 / 4..7,12..15
        }
#pragma unroll
        for (int i = 0; i < 2; ++i)
#pragma unroll
            for (int j = 0; j < 4; ++j)
                acc[i][j] = __builtin_amdgcn_wmma_f32_16x16x32_bf16(
                    false, a[i].v, false, b[j].v, (short)0, acc[i][j], false, false);
    }

    // ---- epilogue ----
#pragma unroll
    for (int i = 0; i < 2; ++i) {
#pragma unroll
        for (int j = 0; j < 4; ++j) {
            int gn = n0 + wn * 64 + j * 16 + lrow;
            if (gn >= N) continue;
            float bv = RELU_BIAS_BF16_OUT ? bias[gn] : 0.0f;
#pragma unroll
            for (int r = 0; r < 8; ++r) {
                int gr = m0 + wm * 32 + i * 16 + r + half * 8;
                if (gr >= M) continue;
                float v = acc[i][j][r];
                if (RELU_BIAS_BF16_OUT) {
                    v += bv;
                    v = v > 0.0f ? v : 0.0f;
                    ((unsigned short*)Cptr)[(size_t)gr * N + gn] = f2bf(v);
                } else {
                    ((float*)Cptr)[(size_t)gr * N + gn] = v;
                }
            }
        }
    }
}

extern "C" void kernel_launch(void* const* d_in, const int* in_sizes, int n_in,
                              void* d_out, int out_size, void* d_ws, size_t ws_size,
                              hipStream_t stream) {
    (void)in_sizes; (void)n_in; (void)out_size; (void)ws_size;
    const float* x   = (const float*)d_in[0];
    const int*   ei  = (const int*)d_in[1];   // [2, E]: row = sources, col = targets
    const float* W1  = (const float*)d_in[2];
    const float* b1  = (const float*)d_in[3];
    const float* W2  = (const float*)d_in[4];
    const float* b2  = (const float*)d_in[5];
    float*       out = (float*)d_out;

    const int* row = ei;
    const int* col = ei + N_EDGES;

    // workspace layout (256B aligned)
    char* ws = (char*)d_ws;
    size_t off = 0;
    auto take = [&](size_t bytes) { char* p = ws + off; off = (off + bytes + 255) & ~(size_t)255; return p; };
    float*          dis   = (float*)take((size_t)N_NODES * 4);
    float*          agg   = (float*)take((size_t)N_NODES * IN_DIM * 4);        // A_norm @ X (f32 atomics)
    unsigned short* aggbf = (unsigned short*)take((size_t)N_NODES * IN_DIM * 2);
    unsigned int*   W1p   = (unsigned int*)take((size_t)(IN_DIM / 2) * HID * 4);
    unsigned int*   W2p   = (unsigned int*)take((size_t)(HID / 2) * LAT * 4);
    unsigned short* Hbf   = (unsigned short*)take((size_t)N_NODES * HID * 2);  // relu(agg@W1+b1)
    float*          Mbuf  = (float*)take((size_t)N_NODES * LAT * 4);           // H @ W2

    // 0) pack weights to bf16 pair layout (tiny; L2 resident afterwards)
    {
        int t1 = (IN_DIM / 2) * HID;
        pack_weight_kernel<<<(t1 + 255) / 256, 256, 0, stream>>>(W1, W1p, IN_DIM / 2, HID);
        int t2 = (HID / 2) * LAT;
        pack_weight_kernel<<<(t2 + 255) / 256, 256, 0, stream>>>(W2, W2p, HID / 2, LAT);
    }

    // 1) degrees & normalization
    (void)hipMemsetAsync(dis, 0, (size_t)N_NODES * 4, stream);
    deg_count_kernel<<<(N_EDGES + 255) / 256, 256, 0, stream>>>(col, dis, N_EDGES);
    dis_kernel<<<(N_NODES + 255) / 256, 256, 0, stream>>>(dis, N_NODES);

    // 2) layer-1 aggregation in input dim (A is linear: A(XW) == (AX)W), then cvt to bf16
    {
        const int F4 = IN_DIM / 4;
        int tsl = N_NODES * F4;
        selfloop_kernel<<<(tsl + 255) / 256, 256, 0, stream>>>(x, dis, agg, N_NODES, F4);
        long long te = (long long)N_EDGES * F4;
        edge_scatter_kernel<<<(int)((te + 255) / 256), 256, 0, stream>>>(row, col, dis, x, agg,
                                                                         N_EDGES, F4);
        int t4 = N_NODES * IN_DIM / 4;
        cvt_bf16_kernel<<<(t4 + 255) / 256, 256, 0, stream>>>(agg, aggbf, t4);
    }

    // 3) GEMM1: H = relu(aggbf @ W1 + b1), bf16 out
    {
        dim3 grid((HID + BN - 1) / BN, (N_NODES + BM - 1) / BM);
        gemm_wmma_kernel<true><<<grid, 256, 0, stream>>>(aggbf, W1p, b1, Hbf,
                                                         N_NODES, HID, IN_DIM);
    }

    // 4) GEMM2: M = H @ W2 (bias applied after aggregation, per reference)
    {
        dim3 grid((LAT + BN - 1) / BN, (N_NODES + BM - 1) / BM);
        gemm_wmma_kernel<false><<<grid, 256, 0, stream>>>(Hbf, W2p, nullptr, Mbuf,
                                                          N_NODES, LAT, HID);
    }

    // 5) layer-2 aggregation in LAT dim, then sigmoid(out + b2)
    {
        const int F4 = LAT / 4;
        int tsl = N_NODES * F4;
        selfloop_kernel<<<(tsl + 255) / 256, 256, 0, stream>>>(Mbuf, dis, out, N_NODES, F4);
        long long te = (long long)N_EDGES * F4;
        edge_scatter_kernel<<<(int)((te + 255) / 256), 256, 0, stream>>>(row, col, dis, Mbuf, out,
                                                                         N_EDGES, F4);
        int total = N_NODES * LAT;
        finalize_kernel<<<(total + 255) / 256, 256, 0, stream>>>(out, b2, total, LAT);
    }
}
